// Decoder_87316685128429
// MI455X (gfx1250) — matrix operand order repaired and checked by reading, not compiled, not run on hardware
//
#include <hip/hip_runtime.h>

// ---------------------------------------------------------------------------
// Decoder step for MI455X (gfx1250, wave32).
//  - Attention score GEMMs (87% of FLOPs): v_wmma_f32_16x16x32_bf16, fused
//    tanh/score-reduce into LDS, per-batch blocks (score mats never hit HBM).
//  - Vocab projection (HBM-bound on 103MB Wy): v_wmma_f32_16x16x4_f32, fp32
//    streamed once (bf16 convert would ADD traffic at M=32), branch-free
//    inner loop (clamped-column trick) + global_prefetch on the B stream.
//  - LSTM / readout GEMMs: f32 WMMA, fused multi-operand + bias.
// ---------------------------------------------------------------------------

#define B_   32
#define LQ_  64
#define E_   512
#define NFL_ 400      // NF*FL = 10*40
#define D_   512
#define H_   512
#define A_   1024
#define R_   1024
#define V_   50257
#define FL_  40
#define NEG_ (-1e9f)

typedef __attribute__((ext_vector_type(2)))  float  v2f;
typedef __attribute__((ext_vector_type(8)))  float  v8f;
typedef __attribute__((ext_vector_type(8)))  __bf16 v8bf;
typedef __attribute__((ext_vector_type(16))) __bf16 v16bf;

static __device__ __forceinline__ v16bf cat8(v8bf lo, v8bf hi) {
  return __builtin_shufflevector(lo, hi, 0,1,2,3,4,5,6,7,8,9,10,11,12,13,14,15);
}

// A operand, bf16 16x32 tile, row-major A[.][lda]  (ISA 7.12.2 16-bit A layout:
// lanes 0-15 hold K 0..7 & 16..23 of row M=lane, lanes 16-31 hold K 8..15 & 24..31)
static __device__ __forceinline__ v16bf load_a16x32(const __bf16* A, int lda,
                                                    int m0, int k0, int lane) {
  const int half = lane >> 4, lm = lane & 15;
  const __bf16* p = A + (size_t)(m0 + lm) * lda + k0 + half * 8;
  v8bf lo = *(const v8bf*)p;
  v8bf hi = *(const v8bf*)(p + 16);
  return cat8(lo, hi);
}

// B operand, bf16 32x16 tile, from PRE-TRANSPOSED weights Wt[N][K]
// (lanes 0-15: col N=lane, K 0..7 & 16..23; lanes 16-31: K 8..15 & 24..31)
static __device__ __forceinline__ v16bf load_b32x16_t(const __bf16* Wt, int ldb,
                                                      int n0, int k0, int lane) {
  const int half = lane >> 4, lm = lane & 15;
  const __bf16* p = Wt + (size_t)(n0 + lm) * ldb + k0 + half * 8;
  v8bf lo = *(const v8bf*)p;
  v8bf hi = *(const v8bf*)(p + 16);
  return cat8(lo, hi);
}

// --------------------------- conversion kernels ----------------------------
__global__ void k_cvt(const float* __restrict__ src, __bf16* __restrict__ dst, int n) {
  for (int i = blockIdx.x * blockDim.x + threadIdx.x; i < n; i += gridDim.x * blockDim.x)
    dst[i] = (__bf16)src[i];
}

// W[K][N] fp32 -> Wt[N][K] bf16
__global__ void k_cvt_t(const float* __restrict__ W, __bf16* __restrict__ Wt,
                        int K, int N, int n) {
  for (int i = blockIdx.x * blockDim.x + threadIdx.x; i < n; i += gridDim.x * blockDim.x) {
    int k = i / N, nn = i % N;
    Wt[(size_t)nn * K + k] = (__bf16)W[(size_t)k * N + nn];
  }
}

// -------------------- generic fused f32-WMMA GEMM (M=32) -------------------
// C[M,N] = sum_p A_p[M,K_p] @ B_p[K_p,N]  + bias0+bias1+bias2   (null => skip)
// 4 waves per block, one 16x16 output tile per wave.
// Branch-free inner loop: out-of-range lanes load a clamped column; their
// accumulators are garbage but never stored (each lane's B col only feeds
// that lane's own output column).
__global__ void __launch_bounds__(128)
k_gemm_f32(const float* A0, const float* B0, int K0,
           const float* A1, const float* B1, int K1,
           const float* A2, const float* B2, int K2,
           const float* bias0, const float* bias1, const float* bias2,
           float* C, int M, int N, int NT) {
  const int lane = threadIdx.x & 31;
  const int wave = threadIdx.x >> 5;
  const int half = lane >> 4, lm = lane & 15;
  const int nt = blockIdx.x * 4 + wave;
  if (nt >= NT) return;                      // wave-uniform
  const int n0 = nt * 16;
  const int m0 = blockIdx.y * 16;
  const int col = n0 + lm;
  const bool cok = col < N;
  const int colc = cok ? col : N - 1;        // clamp: garbage never stored
  v8f acc = {};
  const float* As[3] = {A0, A1, A2};
  const float* Bs[3] = {B0, B1, B2};
  const int    Ks[3] = {K0, K1, K2};
  for (int p = 0; p < 3; ++p) {
    if (!As[p]) continue;
    const int K = Ks[p];
    const float* Arow = As[p] + (size_t)(m0 + lm) * K + 2 * half;
    const float* Bcol = Bs[p] + (size_t)(2 * half) * N + colc;
    for (int k0 = 0; k0 < K; k0 += 4) {
      // f32 A 16x4: lanes 0-15 -> K k0,k0+1 ; lanes 16-31 -> K k0+2,k0+3
      v2f a, b;
      a.x = Arow[0];
      a.y = Arow[1];
      b.x = Bcol[0];
      b.y = Bcol[N];
      __builtin_prefetch(Bcol + (size_t)16 * N, 0, 0);   // global_prefetch_b8
      acc = __builtin_amdgcn_wmma_f32_16x16x4_f32(false, a, false, b,
                                                  (short)0, acc, false, false);
      Arow += 4;
      Bcol += (size_t)4 * N;
    }
  }
  if (cok) {
    float bsum = 0.f;
    if (bias0) bsum += bias0[col];
    if (bias1) bsum += bias1[col];
    if (bias2) bsum += bias2[col];
#pragma unroll
    for (int r = 0; r < 8; ++r) {
      const int row = m0 + half * 8 + r;     // f32 C/D layout (ISA 7.12.2)
      C[(size_t)row * N + col] = acc[r] + bsum;
    }
  }
}

// ------------------- fused question attention (per batch) ------------------
__global__ void __launch_bounds__(256)
k_qattn(const __bf16* __restrict__ bq_bf, const float* __restrict__ bq,
        const __bf16* __restrict__ wq_t, const float* __restrict__ dq,
        const float* __restrict__ vq, const unsigned char* __restrict__ qmask,
        float* __restrict__ out_qlog, float* __restrict__ out_qvec,
        float* __restrict__ ws_qvec) {
  const int b = blockIdx.x;
  const int tid = threadIdx.x;
  const int wave = tid >> 5, lane = tid & 31;
  const int half = lane >> 4, lm = lane & 15;
  __shared__ float s_score[LQ_];
  __shared__ float s_w[LQ_];
  if (tid < LQ_) s_score[tid] = 0.f;
  __syncthreads();

  const __bf16* A = bq_bf + (size_t)b * LQ_ * E_;
  for (int t = wave; t < (LQ_ / 16) * (A_ / 16); t += 8) {   // 256 tiles, 8 waves
    const int m0 = (t >> 6) * 16;
    const int n0 = (t & 63) * 16;
    v8f acc = {};
    for (int k0 = 0; k0 < E_; k0 += 32) {
      v16bf av = load_a16x32(A, E_, m0, k0, lane);
      v16bf bv = load_b32x16_t(wq_t, E_, n0, k0, lane);
      acc = __builtin_amdgcn_wmma_f32_16x16x32_bf16(false, av, false, bv,
                                                    (short)0, acc, false, false);
    }
    const int col = n0 + lm;
    const float dqa = dq[(size_t)b * A_ + col];
    const float vqa = vq[col];
#pragma unroll
    for (int r = 0; r < 8; ++r) {
      const int row = m0 + half * 8 + r;
      atomicAdd(&s_score[row], tanhf(acc[r] + dqa) * vqa);   // ds_add_f32
    }
  }
  __syncthreads();

  if (tid < LQ_) {
    const float sc = s_score[tid];
    const float lg = qmask[(size_t)b * LQ_ + tid] ? sc : NEG_;
    out_qlog[(size_t)b * LQ_ + tid] = lg;
    s_w[tid] = lg;
  }
  __syncthreads();
  if (tid == 0) {
    float mx = -3.4e38f;
    for (int l = 0; l < LQ_; ++l) mx = fmaxf(mx, s_w[l]);
    float s = 0.f;
    for (int l = 0; l < LQ_; ++l) { float e = __expf(s_w[l] - mx); s_w[l] = e; s += e; }
    const float inv = 1.f / s;
    for (int l = 0; l < LQ_; ++l) s_w[l] *= inv;
  }
  __syncthreads();
  for (int e = tid; e < E_; e += 256) {
    float acc = 0.f;
    for (int l = 0; l < LQ_; ++l)
      acc += s_w[l] * bq[((size_t)b * LQ_ + l) * E_ + e];
    out_qvec[(size_t)b * E_ + e] = acc;
    ws_qvec[(size_t)b * E_ + e] = acc;
  }
}

// --------------------- fused facts attention (per batch) -------------------
__global__ void __launch_bounds__(256)
k_fattn(const __bf16* __restrict__ fe_bf, const float* __restrict__ fe,
        const __bf16* __restrict__ wf_t, const float* __restrict__ df,
        const float* __restrict__ vf, const unsigned char* __restrict__ fmask,
        float* __restrict__ ws_fvec) {
  const int b = blockIdx.x;
  const int tid = threadIdx.x;
  const int wave = tid >> 5, lane = tid & 31;
  const int half = lane >> 4, lm = lane & 15;
  __shared__ float s_score[NFL_];
  __shared__ float s_tmp[NFL_];
  __shared__ float s_rv[256];
  __shared__ int   s_ri[256];
  __shared__ int   s_idx[FL_];
  __shared__ float s_w[FL_];
  for (int i = tid; i < NFL_; i += 256) s_score[i] = 0.f;
  __syncthreads();

  const __bf16* A = fe_bf + (size_t)b * NFL_ * D_;
  for (int t = wave; t < (NFL_ / 16) * (A_ / 16); t += 8) {  // 1600 tiles, 8 waves
    const int m0 = (t >> 6) * 16;
    const int n0 = (t & 63) * 16;
    v8f acc = {};
    for (int k0 = 0; k0 < D_; k0 += 32) {
      v16bf av = load_a16x32(A, D_, m0, k0, lane);
      v16bf bv = load_b32x16_t(wf_t, D_, n0, k0, lane);
      acc = __builtin_amdgcn_wmma_f32_16x16x32_bf16(false, av, false, bv,
                                                    (short)0, acc, false, false);
    }
    const int col = n0 + lm;
    const float dfa = df[(size_t)b * A_ + col];
    const float vfa = vf[col];
#pragma unroll
    for (int r = 0; r < 8; ++r) {
      const int row = m0 + half * 8 + r;
      atomicAdd(&s_score[row], tanhf(acc[r] + dfa) * vfa);
    }
  }
  __syncthreads();

  for (int i = tid; i < NFL_; i += 256)
    s_tmp[i] = fmask[(size_t)b * NFL_ + i] ? s_score[i] : NEG_;
  __syncthreads();

  // top-40 of 400 via 40 parallel argmax passes (all in LDS)
  for (int k = 0; k < FL_; ++k) {
    float best = -3.4e38f; int bi = 0;
    for (int i = tid; i < NFL_; i += 256)
      if (s_tmp[i] > best) { best = s_tmp[i]; bi = i; }
    s_rv[tid] = best; s_ri[tid] = bi;
    __syncthreads();
    for (int off = 128; off > 0; off >>= 1) {
      if (tid < off && s_rv[tid + off] > s_rv[tid]) {
        s_rv[tid] = s_rv[tid + off]; s_ri[tid] = s_ri[tid + off];
      }
      __syncthreads();
    }
    if (tid == 0) { s_idx[k] = s_ri[0]; s_w[k] = s_rv[0]; s_tmp[s_ri[0]] = -3.4e38f; }
    __syncthreads();
  }
  if (tid == 0) {
    float mx = -3.4e38f;
    for (int k = 0; k < FL_; ++k) mx = fmaxf(mx, s_w[k]);
    float s = 0.f;
    for (int k = 0; k < FL_; ++k) { float e = __expf(s_w[k] - mx); s_w[k] = e; s += e; }
    const float inv = 1.f / s;
    for (int k = 0; k < FL_; ++k) s_w[k] *= inv;
  }
  __syncthreads();
  for (int e = tid; e < D_; e += 256) {
    float acc = 0.f;
    for (int k = 0; k < FL_; ++k)
      acc += s_w[k] * fe[((size_t)b * NFL_ + s_idx[k]) * D_ + e];
    ws_fvec[(size_t)b * D_ + e] = acc;
  }
}

// ------------------------------ small kernels ------------------------------
__global__ void k_build_x(const float* __restrict__ prev, const float* __restrict__ fv,
                          const float* __restrict__ qv, float* __restrict__ x, int n) {
  const int t = blockIdx.x * blockDim.x + threadIdx.x;
  if (t >= n) return;
  const int XW = 2 * E_ + D_;
  const int b = t / XW, j = t % XW;
  float v;
  if (j < E_)            v = prev[(size_t)b * E_ + j];
  else if (j < E_ + D_)  v = fv[(size_t)b * D_ + (j - E_)];
  else                   v = qv[(size_t)b * E_ + (j - E_ - D_)];
  x[t] = v;
}

__global__ void k_lstm(const float* __restrict__ z, const float* __restrict__ c0,
                       float* __restrict__ out_h, float* __restrict__ out_c,
                       float* __restrict__ ws_h) {
  const int t = blockIdx.x * blockDim.x + threadIdx.x;
  if (t >= B_ * H_) return;
  const int b = t / H_, hh = t % H_;
  const float* zb = z + (size_t)b * 4 * H_;
  const float zi = zb[hh], zf = zb[H_ + hh], zg = zb[2 * H_ + hh], zo = zb[3 * H_ + hh];
  const float si = 1.f / (1.f + __expf(-zi));
  const float sf = 1.f / (1.f + __expf(-zf));
  const float so = 1.f / (1.f + __expf(-zo));
  const float c = sf * c0[t] + si * tanhf(zg);
  const float h = so * tanhf(c);
  out_c[t] = c; out_h[t] = h; ws_h[t] = h;
}

__global__ void k_maxout(const float* __restrict__ r, float* __restrict__ m) {
  const int t = blockIdx.x * blockDim.x + threadIdx.x;
  if (t >= B_ * (R_ / 2)) return;
  const int b = t / (R_ / 2), j = t % (R_ / 2);
  const float a0 = r[(size_t)b * R_ + 2 * j];
  const float a1 = r[(size_t)b * R_ + 2 * j + 1];
  m[t] = fmaxf(a0, a1);
}

// ------------------------------- launcher ----------------------------------
extern "C" void kernel_launch(void* const* d_in, const int* in_sizes, int n_in,
                              void* d_out, int out_size, void* d_ws, size_t ws_size,
                              hipStream_t stream) {
  (void)in_sizes; (void)n_in; (void)out_size; (void)ws_size;
  const float* bq        = (const float*)d_in[0];
  const float* fe        = (const float*)d_in[1];
  const float* h0        = (const float*)d_in[2];
  const float* c0        = (const float*)d_in[3];
  const float* prev      = (const float*)d_in[4];
  const float* Wq_enc    = (const float*)d_in[5];
  const float* Wq_dec    = (const float*)d_in[6];
  const float* vq        = (const float*)d_in[7];
  const float* Wf_enc    = (const float*)d_in[8];
  const float* Wf_dec    = (const float*)d_in[9];
  const float* vf        = (const float*)d_in[10];
  const float* lstm_kern = (const float*)d_in[11];
  const float* lstm_rec  = (const float*)d_in[12];
  const float* lstm_bias = (const float*)d_in[13];
  const float* Wr        = (const float*)d_in[14];
  const float* br        = (const float*)d_in[15];
  const float* Ur        = (const float*)d_in[16];
  const float* bu        = (const float*)d_in[17];
  const float* Vr        = (const float*)d_in[18];
  const float* bv        = (const float*)d_in[19];
  const float* Wy        = (const float*)d_in[20];
  const float* by        = (const float*)d_in[21];
  const unsigned char* qmask = (const unsigned char*)d_in[22];
  const unsigned char* fmask = (const unsigned char*)d_in[23];

  float* out        = (float*)d_out;
  float* out_logits = out;                              // [32, 50257]
  float* out_h      = out_logits + (size_t)B_ * V_;     // [32, 512]
  float* out_c      = out_h + (size_t)B_ * H_;          // [32, 512]
  float* out_qvec   = out_c + (size_t)B_ * H_;          // [32, 512]
  float* out_qlog   = out_qvec + (size_t)B_ * E_;       // [32, 64]

  char* ws = (char*)d_ws;
  size_t off = 0;
  auto alloc = [&](size_t bytes) -> void* {
    void* p = (void*)(ws + off);
    off += (bytes + 255) & ~(size_t)255;
    return p;
  };
  __bf16* bq_bf = (__bf16*)alloc((size_t)B_ * LQ_ * E_ * 2);
  __bf16* wq_t  = (__bf16*)alloc((size_t)A_ * E_ * 2);
  __bf16* fe_bf = (__bf16*)alloc((size_t)B_ * NFL_ * D_ * 2);
  __bf16* wf_t  = (__bf16*)alloc((size_t)A_ * D_ * 2);
  float*  dq    = (float*)alloc((size_t)B_ * A_ * 4);
  float*  df    = (float*)alloc((size_t)B_ * A_ * 4);
  float*  qv    = (float*)alloc((size_t)B_ * E_ * 4);
  float*  fv    = (float*)alloc((size_t)B_ * D_ * 4);
  float*  xb    = (float*)alloc((size_t)B_ * (2 * E_ + D_) * 4);
  float*  zb    = (float*)alloc((size_t)B_ * 4 * H_ * 4);
  float*  hb    = (float*)alloc((size_t)B_ * H_ * 4);
  float*  rb    = (float*)alloc((size_t)B_ * R_ * 4);
  float*  mb    = (float*)alloc((size_t)B_ * (R_ / 2) * 4);

  int n;
  n = B_ * LQ_ * E_;
  k_cvt<<<(n + 255) / 256, 256, 0, stream>>>(bq, bq_bf, n);
  n = B_ * NFL_ * D_;
  k_cvt<<<(n + 255) / 256, 256, 0, stream>>>(fe, fe_bf, n);
  n = E_ * A_;
  k_cvt_t<<<(n + 255) / 256, 256, 0, stream>>>(Wq_enc, wq_t, E_, A_, n);
  n = D_ * A_;
  k_cvt_t<<<(n + 255) / 256, 256, 0, stream>>>(Wf_enc, wf_t, D_, A_, n);

  // dq = h0 @ Wq_dec ; df = h0 @ Wf_dec    (M=32, K=512, N=1024)
  {
    const int NT = A_ / 16;
    k_gemm_f32<<<dim3((NT + 3) / 4, B_ / 16), 128, 0, stream>>>(
        h0, Wq_dec, H_, nullptr, nullptr, 0, nullptr, nullptr, 0,
        nullptr, nullptr, nullptr, dq, B_, A_, NT);
    k_gemm_f32<<<dim3((NT + 3) / 4, B_ / 16), 128, 0, stream>>>(
        h0, Wf_dec, H_, nullptr, nullptr, 0, nullptr, nullptr, 0,
        nullptr, nullptr, nullptr, df, B_, A_, NT);
  }

  k_qattn<<<B_, 256, 0, stream>>>(bq_bf, bq, wq_t, dq, vq, qmask,
                                  out_qlog, out_qvec, qv);
  k_fattn<<<B_, 256, 0, stream>>>(fe_bf, fe, wf_t, df, vf, fmask, fv);

  n = B_ * (2 * E_ + D_);
  k_build_x<<<(n + 255) / 256, 256, 0, stream>>>(prev, fv, qv, xb, n);

  // z = x @ lstm_kernel + h0 @ lstm_rec + bias   (N=2048)
  {
    const int NT = (4 * H_) / 16;
    k_gemm_f32<<<dim3((NT + 3) / 4, B_ / 16), 128, 0, stream>>>(
        xb, lstm_kern, 2 * E_ + D_, h0, lstm_rec, H_, nullptr, nullptr, 0,
        lstm_bias, nullptr, nullptr, zb, B_, 4 * H_, NT);
  }

  n = B_ * H_;
  k_lstm<<<(n + 255) / 256, 256, 0, stream>>>(zb, c0, out_h, out_c, hb);

  // r = h@Wr + x@Ur + q_vec@Vr + (br+bu+bv)     (N=1024)
  {
    const int NT = R_ / 16;
    k_gemm_f32<<<dim3((NT + 3) / 4, B_ / 16), 128, 0, stream>>>(
        hb, Wr, H_, xb, Ur, 2 * E_ + D_, qv, Vr, E_,
        br, bu, bv, rb, B_, R_, NT);
  }

  n = B_ * (R_ / 2);
  k_maxout<<<(n + 255) / 256, 256, 0, stream>>>(rb, mb);

  // logits = m @ Wy + by   (N=50257, HBM-bound: fp32 streamed once)
  {
    const int NT = (V_ + 15) / 16;
    k_gemm_f32<<<dim3((NT + 3) / 4, B_ / 16), 128, 0, stream>>>(
        mb, Wy, R_ / 2, nullptr, nullptr, 0, nullptr, nullptr, 0,
        by, nullptr, nullptr, out_logits, B_, V_, NT);
  }
}
